// NNGuideCriterion_10634339025064
// MI455X (gfx1250) — matrix operand order repaired and checked by reading, not compiled, not run on hardware
//
#include <hip/hip_runtime.h>

// ---------------------------------------------------------------------------
// NNGuide criterion for MI455X (gfx1250, wave32, WMMA).
// Shapes fixed by the reference: Q=4096, D=768, N=100000, C=1000, K=100.
// ---------------------------------------------------------------------------

typedef __attribute__((ext_vector_type(16))) __bf16 v16bf;
typedef __attribute__((ext_vector_type(8)))  float  v8f;

constexpr int Q_DIM  = 4096;
constexpr int D_DIM  = 768;
constexpr int N_BANK = 100000;
constexpr int C_DIM  = 1000;
constexpr int KSEL   = 100;

constexpr int MT = 128;      // query rows per workgroup
constexpr int NT = 64;       // bank cols per workgroup
constexpr int KC = 32;       // K-step (bf16 WMMA depth)
constexpr int NC = 5120;     // bank columns per chunk (sims scratch = 84MB, L2-resident)
constexpr int NCHUNK = (N_BANK + NC - 1) / NC;  // 20

// ---------------- wave32 reductions -----------------
__device__ __forceinline__ float wred_sum(float v) {
#pragma unroll
  for (int o = 16; o; o >>= 1) v += __shfl_xor(v, o, 32);
  return v;
}
__device__ __forceinline__ int wred_sumi(int v) {
#pragma unroll
  for (int o = 16; o; o >>= 1) v += __shfl_xor(v, o, 32);
  return v;
}
__device__ __forceinline__ float wred_max(float v) {
#pragma unroll
  for (int o = 16; o; o >>= 1) v = fmaxf(v, __shfl_xor(v, o, 32));
  return v;
}

// ---------------- bf16 hi/lo split helpers -----------------
__device__ __forceinline__ unsigned bf16_rne(float f) {
  unsigned u = __float_as_uint(f);
  return (u + 0x7FFFu + ((u >> 16) & 1u)) >> 16;
}
// pack two floats into (hi-dword, lo-dword) of bf16 pairs; x in [15:0], y in [31:16]
__device__ __forceinline__ void split2(float x, float y, unsigned& hid, unsigned& lod) {
  unsigned hx = bf16_rne(x), hy = bf16_rne(y);
  float rx = x - __uint_as_float(hx << 16);
  float ry = y - __uint_as_float(hy << 16);
  unsigned lx = bf16_rne(rx), ly = bf16_rne(ry);
  hid = hx | (hy << 16);
  lod = lx | (ly << 16);
}

// ---------------- order-preserving float<->uint keys -----------------
__device__ __forceinline__ unsigned fkey(float f) {
  unsigned u = __float_as_uint(f);
  return (u & 0x80000000u) ? ~u : (u | 0x80000000u);
}
__device__ __forceinline__ float funkey(unsigned k) {
  unsigned u = (k & 0x80000000u) ? (k ^ 0x80000000u) : ~k;
  return __uint_as_float(u);
}

// ---------------------------------------------------------------------------
// Kernel 1: bank scale  s[n] = logsumexp(bank_logits[n,:]) / ||bank_feas[n,:]||
// One wave per row; 8 waves per block.
// ---------------------------------------------------------------------------
__global__ __launch_bounds__(256) void bank_stats_kernel(
    const float* __restrict__ feas, const float* __restrict__ logits,
    float* __restrict__ sbank) {
  int wave = threadIdx.x >> 5, lane = threadIdx.x & 31;
  int r = blockIdx.x * 8 + wave;
  if (r >= N_BANK) return;
  const float* lr = logits + (size_t)r * C_DIM;
  float m = -3.0e38f;
  for (int i = lane; i < C_DIM; i += 32) m = fmaxf(m, lr[i]);
  m = wred_max(m);
  float s = 0.f;
  for (int i = lane; i < C_DIM; i += 32) s += __expf(lr[i] - m);
  s = wred_sum(s);
  const float* fr = feas + (size_t)r * D_DIM;
  float n2 = 0.f;
  for (int i = lane; i < D_DIM; i += 32) { float x = fr[i]; n2 += x * x; }
  n2 = wred_sum(n2);
  if (lane == 0) sbank[r] = (m + __logf(s)) * rsqrtf(n2);
}

// ---------------------------------------------------------------------------
// Kernel 2: query stats  energy[q] = logsumexp(logit[q,:]), qsc[q] = 1/||feature[q,:]||
// ---------------------------------------------------------------------------
__global__ __launch_bounds__(256) void query_stats_kernel(
    const float* __restrict__ feature, const float* __restrict__ logit,
    float* __restrict__ qsc, float* __restrict__ energy) {
  int wave = threadIdx.x >> 5, lane = threadIdx.x & 31;
  int q = blockIdx.x * 8 + wave;
  if (q >= Q_DIM) return;
  const float* lr = logit + (size_t)q * C_DIM;
  float m = -3.0e38f;
  for (int i = lane; i < C_DIM; i += 32) m = fmaxf(m, lr[i]);
  m = wred_max(m);
  float s = 0.f;
  for (int i = lane; i < C_DIM; i += 32) s += __expf(lr[i] - m);
  s = wred_sum(s);
  const float* fr = feature + (size_t)q * D_DIM;
  float n2 = 0.f;
  for (int i = lane; i < D_DIM; i += 32) { float x = fr[i]; n2 += x * x; }
  n2 = wred_sum(n2);
  if (lane == 0) { energy[q] = m + __logf(s); qsc[q] = rsqrtf(n2); }
}

// ---------------------------------------------------------------------------
// Kernel 3: WMMA GEMM chunk. Computes sims[q, ncol] for ncol in [chunk, chunk+NC)
// using bf16 hi/lo split (3x v_wmma_f32_16x16x32_bf16 per 16x16x32 tile).
// WG = 256 threads = 8 waves; each wave owns one 16-row block x 4 col blocks.
// LDS tiles are stored directly in WMMA per-lane operand order:
//   A (16x32, 16-bit): lanes 0-15 hold K{0..7,16..23}, lanes 16-31 K{8..15,24..31}
//   B (32x16, 16-bit): lanes 0-15 hold K{0..7,16..23}, lanes 16-31 K{8..15,24..31} (same as A)
// so each lane reads its v16bf with two ds_load_b128.
// ---------------------------------------------------------------------------
__global__ __launch_bounds__(256) void gemm_kernel(
    const float* __restrict__ feature, const float* __restrict__ bank_feas,
    const float* __restrict__ qsc, const float* __restrict__ sbank,
    float* __restrict__ sims, int chunk_base) {
  __shared__ unsigned Ahi[(MT / 16) * 32 * 8];  // 8 KB
  __shared__ unsigned Alo[(MT / 16) * 32 * 8];  // 8 KB
  __shared__ unsigned Bhi[(NT / 16) * 32 * 8];  // 4 KB
  __shared__ unsigned Blo[(NT / 16) * 32 * 8];  // 4 KB

  const int tid = threadIdx.x;
  const int wave = tid >> 5, lane = tid & 31;
  const int q0 = blockIdx.y * MT;
  const int ncol0 = blockIdx.x * NT;       // column within chunk
  const int n0 = chunk_base + ncol0;       // absolute bank row

  // Hoist per-thread cooperative-load assignments (k-independent).
  int arow[8], aoff[8];
  float asc[8];
#pragma unroll
  for (int i = 0; i < 8; i++) {
    int idx = tid + 256 * i;               // 2048 dword slots: 128 rows x 16 kpairs
    int row = idx >> 4, p = idx & 15;
    arow[i] = row;
    asc[i] = qsc[q0 + row];
    int blk = row >> 4, ml = row & 15;
    int lsel = (p & 4) ? (16 + ml) : ml;
    int j = (p & 3) + ((p & 8) ? 4 : 0);
    aoff[i] = (blk * 32 + lsel) * 8 + j;
  }
  int bng[4], boff[4];
  float bsc[4];
#pragma unroll
  for (int i = 0; i < 4; i++) {
    int idx = tid + 256 * i;               // 1024 dword slots: 64 n-rows x 16 kpairs
    int nl = idx >> 4, p = idx & 15;
    int ng = n0 + nl;
    bng[i] = ng;
    bsc[i] = (ng < N_BANK) ? sbank[ng] : 0.0f;
    int nblk = nl >> 4, nloc = nl & 15;
    int lsel = (p & 4) ? (16 + nloc) : nloc;
    boff[i] = (nblk * 32 + lsel) * 8 + (p & 3) + ((p & 8) ? 4 : 0);
  }

  v8f acc[4] = {};

#pragma unroll 1
  for (int k0 = 0; k0 < D_DIM; k0 += KC) {
    // Issue global loads first (contiguous float2 along D -> coalesced).
    float2 va[8], vb[4];
#pragma unroll
    for (int i = 0; i < 8; i++) {
      int p = (tid + 256 * i) & 15;
      va[i] = *(const float2*)(feature + (size_t)(q0 + arow[i]) * D_DIM + k0 + 2 * p);
    }
#pragma unroll
    for (int i = 0; i < 4; i++) {
      int p = (tid + 256 * i) & 15;
      float2 v = make_float2(0.f, 0.f);
      if (bng[i] < N_BANK)
        v = *(const float2*)(bank_feas + (size_t)bng[i] * D_DIM + k0 + 2 * p);
      vb[i] = v;
    }
    __syncthreads();  // previous iteration's DS reads done
#pragma unroll
    for (int i = 0; i < 8; i++) {
      unsigned h, l;
      split2(va[i].x * asc[i], va[i].y * asc[i], h, l);
      Ahi[aoff[i]] = h;
      Alo[aoff[i]] = l;
    }
#pragma unroll
    for (int i = 0; i < 4; i++) {
      unsigned h, l;
      split2(vb[i].x * bsc[i], vb[i].y * bsc[i], h, l);
      Bhi[boff[i]] = h;
      Blo[boff[i]] = l;
    }
    __syncthreads();

    const v16bf ah = *(const v16bf*)&Ahi[(wave * 32 + lane) * 8];
    const v16bf al = *(const v16bf*)&Alo[(wave * 32 + lane) * 8];
#pragma unroll
    for (int nb = 0; nb < 4; nb++) {
      const v16bf bh = *(const v16bf*)&Bhi[(nb * 32 + lane) * 8];
      const v16bf bl = *(const v16bf*)&Blo[(nb * 32 + lane) * 8];
      // x*y ~= hi*hi + hi*lo + lo*hi  (drop lo*lo: ~2^-16 relative error)
      acc[nb] = __builtin_amdgcn_wmma_f32_16x16x32_bf16(false, ah, false, bh,
                                                        (short)0, acc[nb], false, false);
      acc[nb] = __builtin_amdgcn_wmma_f32_16x16x32_bf16(false, ah, false, bl,
                                                        (short)0, acc[nb], false, false);
      acc[nb] = __builtin_amdgcn_wmma_f32_16x16x32_bf16(false, al, false, bh,
                                                        (short)0, acc[nb], false, false);
    }
  }

  // C/D layout: VGPR j -> M = j (lanes 0-15) or 8+j (lanes 16-31); N = lane%16.
  const int qrow = (lane < 16) ? 0 : 8;
  const int nl = lane & 15;
#pragma unroll
  for (int nb = 0; nb < 4; nb++) {
    int ncol = ncol0 + nb * 16 + nl;
    bool valid = (chunk_base + ncol) < N_BANK;
#pragma unroll
    for (int j = 0; j < 8; j++) {
      int q = q0 + wave * 16 + qrow + j;
      float vv = valid ? acc[nb][j] : -3.0e38f;  // pad tail with -inf-ish
      sims[(size_t)q * NC + ncol] = vv;
    }
  }
}

// ---------------------------------------------------------------------------
// Kernel 4: exact per-chunk top-K. One wave per query. Bitwise binary search
// for the K-th largest key over 5120 LDS-resident keys, then compact-write the
// top-K multiset (values > t, padded with copies of t).
// ---------------------------------------------------------------------------
__global__ __launch_bounds__(32) void select_kernel(
    const float* __restrict__ sims, float* __restrict__ cand, int chunk) {
  __shared__ unsigned keys[NC];  // 20 KB
  __shared__ int ctr;
  const int lane = threadIdx.x;
  const int q = blockIdx.x;
  const float* row = sims + (size_t)q * NC;
  for (int i = lane; i < NC; i += 32) keys[i] = fkey(row[i]);
  if (lane == 0) ctr = 0;
  __syncthreads();

  unsigned T = 0;  // max T with count(key >= T) >= K  == K-th largest key
#pragma unroll 1
  for (int b = 31; b >= 0; b--) {
    unsigned trial = T | (1u << b);
    int c = 0;
    for (int i = lane; i < NC; i += 32) c += (keys[i] >= trial);
    c = wred_sumi(c);
    if (c >= KSEL) T = trial;
  }

  float* dst = cand + (size_t)q * (NCHUNK * KSEL) + (size_t)chunk * KSEL;
  for (int i = lane; i < NC; i += 32) {
    unsigned k = keys[i];
    if (k > T) {
      int s = atomicAdd(&ctr, 1);
      dst[s] = funkey(k);
    }
  }
  __syncthreads();
  int g = ctr;               // strictly-greater count, g < K guaranteed
  float t = funkey(T);
  for (int s = g + lane; s < KSEL; s += 32) dst[s] = t;  // fill ties
}

// ---------------------------------------------------------------------------
// Kernel 5: merge 20*K candidates per query -> mean(top-K), score, negate.
// ---------------------------------------------------------------------------
__global__ __launch_bounds__(32) void finalize_kernel(
    const float* __restrict__ cand, const float* __restrict__ energy,
    float* __restrict__ out) {
  constexpr int M = NCHUNK * KSEL;  // 2000
  __shared__ unsigned keys[M];
  const int lane = threadIdx.x;
  const int q = blockIdx.x;
  const float* row = cand + (size_t)q * M;
  for (int i = lane; i < M; i += 32) keys[i] = fkey(row[i]);
  __syncthreads();

  unsigned T = 0;
#pragma unroll 1
  for (int b = 31; b >= 0; b--) {
    unsigned trial = T | (1u << b);
    int c = 0;
    for (int i = lane; i < M; i += 32) c += (keys[i] >= trial);
    c = wred_sumi(c);
    if (c >= KSEL) T = trial;
  }

  float s = 0.f;
  int g = 0;
  for (int i = lane; i < M; i += 32) {
    unsigned k = keys[i];
    if (k > T) { s += funkey(k); g++; }
  }
  s = wred_sum(s);
  g = wred_sumi(g);
  float t = funkey(T);
  float conf = (s + (float)(KSEL - g) * t) * (1.0f / KSEL);
  if (lane == 0) out[q] = -(conf * energy[q]);
}

// ---------------------------------------------------------------------------
// Host-side orchestration (all on `stream`, graph-capture safe).
// ---------------------------------------------------------------------------
extern "C" void kernel_launch(void* const* d_in, const int* in_sizes, int n_in,
                              void* d_out, int out_size, void* d_ws, size_t ws_size,
                              hipStream_t stream) {
  (void)in_sizes; (void)n_in; (void)out_size; (void)ws_size;
  const float* feature     = (const float*)d_in[0];  // [Q, D]
  const float* logit       = (const float*)d_in[1];  // [Q, C]
  const float* bank_feas   = (const float*)d_in[2];  // [N, D]
  const float* bank_logits = (const float*)d_in[3];  // [N, C]
  // d_in[4] = K (==100, baked in as KSEL)
  float* out = (float*)d_out;                        // [Q]

  // Workspace layout (~117 MB total)
  char* w = (char*)d_ws;
  size_t off = 0;
  auto take = [&](size_t n) -> char* {
    char* p = w + off;
    off = (off + n + 255) & ~(size_t)255;
    return p;
  };
  float* sbank  = (float*)take(sizeof(float) * (size_t)N_BANK);
  float* qsc    = (float*)take(sizeof(float) * (size_t)Q_DIM);
  float* energy = (float*)take(sizeof(float) * (size_t)Q_DIM);
  float* cand   = (float*)take(sizeof(float) * (size_t)Q_DIM * NCHUNK * KSEL);  // 32.8 MB
  float* sims   = (float*)take(sizeof(float) * (size_t)Q_DIM * NC);             // 83.9 MB

  bank_stats_kernel<<<(N_BANK + 7) / 8, 256, 0, stream>>>(bank_feas, bank_logits, sbank);
  query_stats_kernel<<<Q_DIM / 8, 256, 0, stream>>>(feature, logit, qsc, energy);

  for (int c = 0; c < NCHUNK; c++) {
    gemm_kernel<<<dim3(NC / NT, Q_DIM / MT), 256, 0, stream>>>(
        feature, bank_feas, qsc, sbank, sims, c * NC);
    select_kernel<<<Q_DIM, 32, 0, stream>>>(sims, cand, c);
  }
  finalize_kernel<<<Q_DIM, 32, 0, stream>>>(cand, energy, out);
}
